// Pointnet_with_transformer_37933151158472
// MI455X (gfx1250) — matrix operand order repaired and checked by reading, not compiled, not run on hardware
//
#include <hip/hip_runtime.h>
#include <hip/hip_bf16.h>

// ---------------------------------------------------------------------------
// MI455X (gfx1250) implementation of PointNet-with-transformer.
// All matrix math mapped to v_wmma_f32_16x16x32_f16 (wave32 WMMA).
// - Fragments are 2 x b128 loads (A: two 16B runs; B: pre-swizzled weights /
//   transposed f16 activations).
// - GEMM/MLP waves run TWO independent accumulator chains to hide the
//   WMMA->WMMA RAW hazard (ISA 7.12.1) between K-steps.
// ---------------------------------------------------------------------------

typedef __attribute__((ext_vector_type(16))) _Float16 v16h;
typedef __attribute__((ext_vector_type(8)))  float    v8f;

#define EPSV 1e-5f

static __device__ __forceinline__ v16h ld_frag2(const _Float16* lo,
                                                const _Float16* hi) {
  union { uint4 u[2]; v16h h; } t;
  t.u[0] = *(const uint4*)lo;   // 16B -> elements 0..7
  t.u[1] = *(const uint4*)hi;   // 16B -> elements 8..15
  return t.h;
}

// A (16x32 f16), source row-major, lda in elements (16B-aligned rows).
// Lanes 0-15: row M=lane, K {0..15}; lanes 16-31: same rows, K {16..31}.
static __device__ __forceinline__ v16h load_a_tile(const _Float16* base, int lda,
                                                   int k0, int lane) {
  const int row = lane & 15;
  const int kh  = (lane >> 4) << 4;                   // 0 or 16
  const _Float16* rp = base + (size_t)row * lda + k0 + kh;
  return ld_frag2(rp, rp + 8);
}

// B fragment where source is stored transposed (element (k,n) at base[n*ld+k]).
static __device__ __forceinline__ v16h load_bt_tile(const _Float16* base, int ld,
                                                    int k0, int n0, int lane) {
  const int n  = n0 + (lane & 15);
  const int kb = k0 + ((lane >> 4) << 4);
  const _Float16* p = base + (size_t)n * ld + kb;
  return ld_frag2(p, p + 8);
}

// B fragment from pre-swizzled weights: [kc][nt][lane][16] contiguous.
static __device__ __forceinline__ v16h load_bsw_tile(const _Float16* Wsw,
                                                     int Ntiles, int kc, int nt,
                                                     int lane) {
  const _Float16* p = Wsw + (((size_t)(kc * Ntiles + nt) * 32) + lane) * 16;
  return ld_frag2(p, p + 8);
}

static __device__ __forceinline__ v8f wmma_f16(v16h a, v16h b, v8f c) {
  return __builtin_amdgcn_wmma_f32_16x16x32_f16(false, a, false, b,
                                                (short)0, c, false, false);
}

// ---------------------------------------------------------------------------
// 1) Weight prep: f32 -> f16, pad K to mult 32 / N to mult 16, and swizzle
//    into WMMA B-fragment order so every fragment load is 2 x b128.
// ---------------------------------------------------------------------------
struct WDesc { const float* src; int K, N, Kp, Np, off; };
struct WPack { WDesc w[10]; };

__global__ __launch_bounds__(256) void prep_weights_kernel(WPack p, _Float16* dst) {
  const WDesc d = p.w[blockIdx.x];
  const int Ntiles = d.Np >> 4;
  const int tot = d.Kp * d.Np;
  for (int i = threadIdx.x; i < tot; i += 256) {
    const int e  = i & 15;
    const int l  = (i >> 4) & 31;
    const int t  = i >> 9;
    const int nt = t % Ntiles;
    const int kc = t / Ntiles;
    const int n  = nt * 16 + (l & 15);
    const int k  = kc * 32 + ((l >> 4) << 4) + e;
    const float v = (k < d.K && n < d.N) ? d.src[(size_t)k * d.N + n] : 0.0f;
    dst[d.off + i] = (_Float16)v;
  }
}

// ---------------------------------------------------------------------------
// 2) Farthest point sampling: 1 block per batch, 1024 threads, points in regs.
// ---------------------------------------------------------------------------
__global__ __launch_bounds__(1024) void fps_kernel(const float* __restrict__ pts,
                                                   float* __restrict__ cent,
                                                   float* __restrict__ cent_out) {
  const int b = blockIdx.x;
  const int tid = threadIdx.x;
  const int lane = tid & 31, wave = tid >> 5;
  __shared__ float cx_s, cy_s, cz_s;
  __shared__ int far_s;
  __shared__ float wval[32];
  __shared__ int   widx[32];

  float px[16], py[16], pz[16], dist[16];
  const float* pb = pts + (size_t)b * 16384 * 3;
#pragma unroll
  for (int i = 0; i < 16; ++i) {
    const int p = tid + i * 1024;
    px[i] = pb[p * 3 + 0]; py[i] = pb[p * 3 + 1]; pz[i] = pb[p * 3 + 2];
    dist[i] = 1e10f;
  }
  if (tid == 0) far_s = 0;
  __syncthreads();

  for (int s = 0; s < 1024; ++s) {
    if (tid == 0) {
      const int f = far_s;
      const float cx = pb[f * 3 + 0], cy = pb[f * 3 + 1], cz = pb[f * 3 + 2];
      cx_s = cx; cy_s = cy; cz_s = cz;
      const size_t o = ((size_t)b * 1024 + s) * 3;
      cent[o + 0] = cx; cent[o + 1] = cy; cent[o + 2] = cz;
      cent_out[o + 0] = cx; cent_out[o + 1] = cy; cent_out[o + 2] = cz;
    }
    __syncthreads();
    const float cx = cx_s, cy = cy_s, cz = cz_s;
    float bv = -1.0f; int bi = 0;
#pragma unroll
    for (int i = 0; i < 16; ++i) {
      const float dx = px[i] - cx, dy = py[i] - cy, dz = pz[i] - cz;
      dist[i] = fminf(dist[i], dx * dx + dy * dy + dz * dz);
      const int p = tid + i * 1024;
      if (dist[i] > bv || (dist[i] == bv && p < bi)) { bv = dist[i]; bi = p; }
    }
    for (int d = 1; d < 32; d <<= 1) {            // wave32 argmax
      const float ov = __shfl_xor(bv, d, 32);
      const int   oi = __shfl_xor(bi, d, 32);
      if (ov > bv || (ov == bv && oi < bi)) { bv = ov; bi = oi; }
    }
    if (lane == 0) { wval[wave] = bv; widx[wave] = bi; }
    __syncthreads();
    if (wave == 0) {
      bv = wval[lane]; bi = widx[lane];
      for (int d = 1; d < 32; d <<= 1) {
        const float ov = __shfl_xor(bv, d, 32);
        const int   oi = __shfl_xor(bi, d, 32);
        if (ov > bv || (ov == bv && oi < bi)) { bv = ov; bi = oi; }
      }
      if (lane == 0) far_s = bi;
    }
    __syncthreads();
  }
}

// ---------------------------------------------------------------------------
// 3) Brute-force 32-NN per centroid; distance row lives in LDS (64KB < 320KB).
// ---------------------------------------------------------------------------
__global__ __launch_bounds__(256) void knn_kernel(const float* __restrict__ pts,
                                                  const float* __restrict__ cent,
                                                  float* __restrict__ gxyz) {
  __shared__ float d2[16384];
  __shared__ float wv[8]; __shared__ int wi[8];
  const int q = blockIdx.x;                 // 0..8191
  const int b = q >> 10;
  const int tid = threadIdx.x, lane = tid & 31, wave = tid >> 5;
  const float* pb = pts + (size_t)b * 16384 * 3;
  const float qx = cent[(size_t)q * 3 + 0];
  const float qy = cent[(size_t)q * 3 + 1];
  const float qz = cent[(size_t)q * 3 + 2];
  for (int j = 0; j < 64; ++j) {
    const int p = tid + j * 256;
    const float dx = pb[p * 3 + 0] - qx, dy = pb[p * 3 + 1] - qy, dz = pb[p * 3 + 2] - qz;
    d2[p] = dx * dx + dy * dy + dz * dz;
  }
  __syncthreads();
  for (int kk = 0; kk < 32; ++kk) {
    float bv = 3.3e38f; int bi = 0;
    for (int j = 0; j < 64; ++j) {
      const int p = tid + j * 256;
      const float v = d2[p];
      if (v < bv || (v == bv && p < bi)) { bv = v; bi = p; }
    }
    for (int d = 1; d < 32; d <<= 1) {
      const float ov = __shfl_xor(bv, d, 32);
      const int   oi = __shfl_xor(bi, d, 32);
      if (ov < bv || (ov == bv && oi < bi)) { bv = ov; bi = oi; }
    }
    if (lane == 0) { wv[wave] = bv; wi[wave] = bi; }
    __syncthreads();
    if (tid == 0) {
      float fv = wv[0]; int fi = wi[0];
      for (int w = 1; w < 8; ++w)
        if (wv[w] < fv || (wv[w] == fv && wi[w] < fi)) { fv = wv[w]; fi = wi[w]; }
      const size_t o = ((size_t)q * 32 + kk) * 3;
      gxyz[o + 0] = pb[fi * 3 + 0];
      gxyz[o + 1] = pb[fi * 3 + 1];
      gxyz[o + 2] = pb[fi * 3 + 2];
      d2[fi] = 3.4e38f;
    }
    __syncthreads();
  }
}

// ---------------------------------------------------------------------------
// 4) Fused per-centroid MLP stack + maxpool, all activations LDS-resident.
//    Each wave runs a PAIR of N-tiles with independent accumulators so the
//    K-loop interleaves two WMMA chains (hides D->C hazard NOPs).
// ---------------------------------------------------------------------------
struct MlpLayer { int Kp, Np, Nreal, woff;
                  const float* bias; const float* bng; const float* bnb; };
struct MlpCfg { int k; MlpLayer L[3]; int colOff;
                const float* gxyz; const _Float16* wf16; float* xout; };

__global__ __launch_bounds__(128) void mlp_kernel(MlpCfg cfg) {
  __shared__ _Float16 actA[32 * 128];
  __shared__ _Float16 actB[32 * 128];
  __shared__ _Float16 wlds[128 * 128];   // swizzled fragment order
  __shared__ float b0lds[128], b1lds[128], b2lds[128];
  const int q = blockIdx.x;
  const int tid = threadIdx.x, lane = tid & 31, wave = tid >> 5;
  const float invs = rsqrtf(1.0f + EPSV);

  for (int i = tid; i < 32 * 128; i += 128) actA[i] = (_Float16)0.0f;
  __syncthreads();
  for (int i = tid; i < cfg.k * 3; i += 128) {
    const int r = i / 3, c = i % 3;
    actA[r * 128 + c] = (_Float16)cfg.gxyz[((size_t)q * 32 + r) * 3 + c];
  }
  __syncthreads();

  _Float16* in = actA; _Float16* out = actB;
  const int Mtiles = cfg.k >> 4;

  for (int l = 0; l < 3; ++l) {
    const MlpLayer L = cfg.L[l];
    // vectorized copy of swizzled weights into LDS
    {
      const uint4* src4 = (const uint4*)(cfg.wf16 + L.woff);
      uint4* dst4 = (uint4*)wlds;
      const int n4 = (L.Kp * L.Np) >> 3;
      for (int i = tid; i < n4; i += 128) dst4[i] = src4[i];
    }
    for (int i = tid; i < L.Np; i += 128) {
      b0lds[i] = (i < L.Nreal) ? L.bias[i] : 0.0f;
      b1lds[i] = (i < L.Nreal) ? L.bng[i]  : 0.0f;
      b2lds[i] = (i < L.Nreal) ? L.bnb[i]  : 0.0f;
    }
    for (int i = tid; i < 32 * 128; i += 128) out[i] = (_Float16)0.0f;
    __syncthreads();

    const int Ntiles = L.Np >> 4;          // always even (2/4/6/8)
    const int total  = Mtiles * Ntiles;
    const int Kt     = L.Kp >> 5;
    for (int tt = wave * 2; tt < total; tt += 8) {
      const int mt  = tt / Ntiles;
      const int nt0 = tt % Ntiles;
      const int nt1 = nt0 + 1;             // same mt: Ntiles even, tt even
      v8f acc0 = {}, acc1 = {};
      for (int kc = 0; kc < Kt; ++kc) {
        const v16h a   = load_a_tile(in + mt * 16 * 128, 128, kc * 32, lane);
        const v16h bm0 = load_bsw_tile(wlds, Ntiles, kc, nt0, lane);
        const v16h bm1 = load_bsw_tile(wlds, Ntiles, kc, nt1, lane);
        acc0 = wmma_f16(a, bm0, acc0);
        acc1 = wmma_f16(a, bm1, acc1);
      }
      const int rbase = mt * 16 + ((lane >> 4) << 3);
#pragma unroll
      for (int half = 0; half < 2; ++half) {
        const v8f acc = half ? acc1 : acc0;
        const int col = (half ? nt1 : nt0) * 16 + (lane & 15);
        if (col < L.Nreal) {
#pragma unroll
          for (int i = 0; i < 8; ++i) {
            float y = acc[i] + b0lds[col];
            y = b1lds[col] * y * invs + b2lds[col]; // bn with 1/sqrt(1+eps)
            y = fmaxf(y, 0.0f);                     // relu
            out[(rbase + i) * 128 + col] = (_Float16)y;
          }
        }
      }
    }
    __syncthreads();
    if (l != 2) {                                   // concat gxyz channels
      for (int i = tid; i < cfg.k * 3; i += 128) {
        const int r = i / 3, c = i % 3;
        out[r * 128 + L.Nreal + c] =
            (_Float16)cfg.gxyz[((size_t)q * 32 + r) * 3 + c];
      }
      __syncthreads();
    }
    _Float16* t = in; in = out; out = t;
  }

  // maxpool over neighbors -> feature slice
  const int Nout = cfg.L[2].Nreal;
  for (int c = tid; c < Nout; c += 128) {
    float m = -3.4e38f;
    for (int r = 0; r < cfg.k; ++r) m = fmaxf(m, (float)in[r * 128 + c]);
    cfg.xout[(size_t)q * 256 + cfg.colOff + c] = m;
  }
}

// ---------------------------------------------------------------------------
// 5) Append centroid xyz columns; emit f16 mirror of x and a per-batch
//    transposed copy xhT[b][ch][row] for contiguous attention-V fragments.
// ---------------------------------------------------------------------------
__global__ __launch_bounds__(256) void finalize_features_kernel(
    const float* __restrict__ cent, float* __restrict__ x,
    _Float16* __restrict__ xh, _Float16* __restrict__ xhT) {
  const int tid = threadIdx.x, lane = tid & 31, wave = tid >> 5;
  const size_t row = (size_t)blockIdx.x * 8 + wave;
  const size_t b = row >> 10, s = row & 1023;
#pragma unroll
  for (int j = 0; j < 8; ++j) {
    const int d = lane + j * 32;
    float v;
    if (d >= 253) v = cent[row * 3 + (d - 253)];
    else          v = x[row * 256 + d];
    x[row * 256 + d]  = v;
    xh[row * 256 + d] = (_Float16)v;
    xhT[((b * 256 + d) << 10) + s] = (_Float16)v;
  }
}

// ---------------------------------------------------------------------------
// 6) Generic WMMA GEMM: out = [relu](A(M,K) * Wsw(K,N) + bias).
//    Each wave owns TWO adjacent N-tiles (dual accumulators, shared A frag).
//    WG covers 16 rows x 128 cols; grid = (M/16, N/128).
// ---------------------------------------------------------------------------
__global__ __launch_bounds__(128) void gemm_kernel(
    const _Float16* __restrict__ A, int K,
    const _Float16* __restrict__ Wsw, int N,
    const float* __restrict__ bias, int relu,
    _Float16* __restrict__ outH, float* __restrict__ outF) {
  __shared__ _Float16 Ash[16 * 512];
  const int tid = threadIdx.x, lane = tid & 31, wave = tid >> 5;
  const int m0  = blockIdx.x * 16;
  const int nt0 = (blockIdx.y * 4 + wave) * 2;
  const int nt1 = nt0 + 1;
  const int Ntiles = N >> 4;
  __builtin_prefetch(Wsw + (size_t)nt0 * 512, 0, 3); // pull weight tiles to WGP
  {
    const uint4* src4 = (const uint4*)(A + (size_t)m0 * K);
    uint4* dst4 = (uint4*)Ash;
    const int n4 = (16 * K) >> 3;
    for (int i = tid; i < n4; i += 128) dst4[i] = src4[i];
  }
  __syncthreads();
  v8f acc0 = {}, acc1 = {};
  const int Kt = K >> 5;
  for (int kc = 0; kc < Kt; ++kc) {
    const v16h a   = load_a_tile(Ash, K, kc * 32, lane);
    const v16h bm0 = load_bsw_tile(Wsw, Ntiles, kc, nt0, lane);
    const v16h bm1 = load_bsw_tile(Wsw, Ntiles, kc, nt1, lane);
    acc0 = wmma_f16(a, bm0, acc0);
    acc1 = wmma_f16(a, bm1, acc1);
  }
  const int rb = m0 + ((lane >> 4) << 3);
#pragma unroll
  for (int half = 0; half < 2; ++half) {
    const v8f acc = half ? acc1 : acc0;
    const int col = (half ? nt1 : nt0) * 16 + (lane & 15);
    const float bv = bias[col];
#pragma unroll
    for (int i = 0; i < 8; ++i) {
      float y = acc[i] + bv;
      if (relu) y = fmaxf(y, 0.0f);
      if (outH) outH[(size_t)(rb + i) * N + col] = (_Float16)y;
      if (outF) outF[(size_t)(rb + i) * N + col] = y;
    }
  }
}

// ---------------------------------------------------------------------------
// 7) Flash attention per (batch, head): head_dim 32 == WMMA K. One wave owns a
//    16-query tile, streams 32-key blocks, online softmax via 16-lane xor
//    shuffles, P re-layout through a per-wave LDS tile. V read from xhT so all
//    fragments are contiguous 2 x b128 loads. s0/s1 and o0/o1 are independent
//    WMMA chains.
// ---------------------------------------------------------------------------
__global__ __launch_bounds__(128) void attn_kernel(
    const _Float16* __restrict__ qf, const _Float16* __restrict__ kf,
    const _Float16* __restrict__ vT, float* __restrict__ att) {
  __shared__ _Float16 P[4][16 * 32];
  const int tid = threadIdx.x, lane = tid & 31, wave = tid >> 5;
  const int id = blockIdx.x;                // 1024 blocks = 8b * 8h * 16
  const int b  = id >> 7;
  const int h  = (id >> 4) & 7;
  const int qt = (id & 15) * 4 + wave;      // 0..63
  const int q0 = qt * 16;
  const _Float16* qp  = qf + ((size_t)(b * 1024 + q0) * 256 + h * 32);
  const _Float16* vtp = vT + (((size_t)b * 256 + h * 32) << 10);
  const v16h aq = load_a_tile(qp, 256, 0, lane);
  const float scale = 0.17677669529663687f; // 1/sqrt(32)
  float m[8], lsum[8];
#pragma unroll
  for (int i = 0; i < 8; ++i) { m[i] = -3.0e38f; lsum[i] = 0.0f; }
  v8f o0 = {}, o1 = {};

  for (int kb = 0; kb < 32; ++kb) {
    const _Float16* kp = kf + ((size_t)(b * 1024 + kb * 32) * 256 + h * 32);
    const v16h bk0 = load_bt_tile(kp, 256, 0, 0, lane);
    const v16h bk1 = load_bt_tile(kp, 256, 0, 16, lane);
    v8f z = {};
    const v8f s0 = wmma_f16(aq, bk0, z);
    const v8f s1 = wmma_f16(aq, bk1, z);
    float p0[8], p1[8];
#pragma unroll
    for (int i = 0; i < 8; ++i) {
      const float a0 = s0[i] * scale, a1 = s1[i] * scale;
      float mx = fmaxf(a0, a1);
      for (int d = 1; d < 16; d <<= 1) mx = fmaxf(mx, __shfl_xor(mx, d, 32));
      const float mn = fmaxf(m[i], mx);
      const float e0 = __expf(a0 - mn), e1 = __expf(a1 - mn);
      float ps = e0 + e1;
      for (int d = 1; d < 16; d <<= 1) ps += __shfl_xor(ps, d, 32);
      const float co = __expf(m[i] - mn);
      lsum[i] = lsum[i] * co + ps;
      o0[i] *= co; o1[i] *= co;
      m[i] = mn;
      p0[i] = e0; p1[i] = e1;
    }
    {
      const int col = lane & 15;
      const int rb  = (lane >> 4) << 3;
#pragma unroll
      for (int i = 0; i < 8; ++i) {
        P[wave][(rb + i) * 32 + col]      = (_Float16)p0[i];
        P[wave][(rb + i) * 32 + 16 + col] = (_Float16)p1[i];
      }
    }
    __syncthreads();
    const v16h ap  = load_a_tile(P[wave], 32, 0, lane);
    const v16h bv0 = load_bt_tile(vtp, 1024, kb * 32, 0, lane);
    const v16h bv1 = load_bt_tile(vtp, 1024, kb * 32, 16, lane);
    o0 = wmma_f16(ap, bv0, o0);
    o1 = wmma_f16(ap, bv1, o1);
    __syncthreads();
  }
  const int col = lane & 15;
  const int rb  = q0 + ((lane >> 4) << 3);
#pragma unroll
  for (int i = 0; i < 8; ++i) {
    const float inv = 1.0f / lsum[i];
    att[((size_t)(b * 1024) + rb + i) * 256 + h * 32 + col]      = o0[i] * inv;
    att[((size_t)(b * 1024) + rb + i) * 256 + h * 32 + 16 + col] = o1[i] * inv;
  }
}

// ---------------------------------------------------------------------------
// 8) Residual add + layernorm (256 channels); one wave per row.
// ---------------------------------------------------------------------------
__global__ __launch_bounds__(256) void ln_add_kernel(
    const float* __restrict__ a, const float* __restrict__ r,
    const float* __restrict__ g, const float* __restrict__ bb,
    float* __restrict__ outF, _Float16* __restrict__ outH) {
  const int tid = threadIdx.x, lane = tid & 31, wave = tid >> 5;
  const size_t row = (size_t)blockIdx.x * 8 + wave;
  float v[8];
  float s = 0.0f;
#pragma unroll
  for (int j = 0; j < 8; ++j) {
    const int d = lane + j * 32;
    v[j] = a[row * 256 + d] + r[row * 256 + d];
    s += v[j];
  }
  for (int d = 1; d < 32; d <<= 1) s += __shfl_xor(s, d, 32);
  const float mean = s * (1.0f / 256.0f);
  float vs = 0.0f;
#pragma unroll
  for (int j = 0; j < 8; ++j) { const float t = v[j] - mean; vs += t * t; }
  for (int d = 1; d < 32; d <<= 1) vs += __shfl_xor(vs, d, 32);
  const float inv = rsqrtf(vs * (1.0f / 256.0f) + EPSV);
#pragma unroll
  for (int j = 0; j < 8; ++j) {
    const int d = lane + j * 32;
    const float y = g[d] * (v[j] - mean) * inv + bb[d];
    outF[row * 256 + d] = y;
    if (outH) outH[row * 256 + d] = (_Float16)y;
  }
}

// ---------------------------------------------------------------------------
// Host orchestration
// ---------------------------------------------------------------------------
extern "C" void kernel_launch(void* const* d_in, const int* in_sizes, int n_in,
                              void* d_out, int out_size, void* d_ws, size_t ws_size,
                              hipStream_t stream) {
  (void)in_sizes; (void)n_in; (void)out_size; (void)ws_size;
  const float* pts = (const float*)d_in[0];
  const float* w[37];
  for (int i = 0; i < 37; ++i) w[i] = (const float*)d_in[i];

  char* ws = (char*)d_ws;
  size_t off = 0;
  auto take = [&](size_t bytes) -> void* {
    void* p = ws + off;
    off = (off + bytes + 255) & ~(size_t)255;
    return p;
  };
  _Float16* wf16 = (_Float16*)take(438272 * 2);
  float*    cent = (float*)take((size_t)8 * 1024 * 3 * 4);
  float*    gxyz = (float*)take((size_t)8 * 1024 * 32 * 3 * 4);
  float*    x    = (float*)take((size_t)8192 * 256 * 4);
  _Float16* xh   = (_Float16*)take((size_t)8192 * 256 * 2);
  _Float16* xhT  = (_Float16*)take((size_t)8192 * 256 * 2);
  _Float16* qh   = (_Float16*)take((size_t)8192 * 256 * 2);
  _Float16* kh   = (_Float16*)take((size_t)8192 * 256 * 2);
  float*    att  = (float*)take((size_t)8192 * 256 * 4);   // reused as h2
  float*    mid  = (float*)take((size_t)8192 * 256 * 4);
  _Float16* midh = (_Float16*)take((size_t)8192 * 256 * 2);
  _Float16* h1h  = (_Float16*)take((size_t)8192 * 512 * 2);

  float* feat_out = (float*)d_out;
  float* cent_out = feat_out + (size_t)8192 * 256;

  // padded, fragment-swizzled f16 weight table (offsets in f16 elements)
  WPack pk;
  pk.w[0] = WDesc{w[1],  3,   32,  32,  32,  0};
  pk.w[1] = WDesc{w[5],  35,  64,  64,  64,  1024};
  pk.w[2] = WDesc{w[9],  67,  125, 96,  128, 5120};
  pk.w[3] = WDesc{w[13], 3,   64,  32,  64,  17408};
  pk.w[4] = WDesc{w[17], 67,  96,  96,  96,  19456};
  pk.w[5] = WDesc{w[21], 99,  128, 128, 128, 28672};
  pk.w[6] = WDesc{w[25], 256, 256, 256, 256, 45056};
  pk.w[7] = WDesc{w[27], 256, 256, 256, 256, 110592};
  pk.w[8] = WDesc{w[33], 256, 512, 256, 512, 176128};
  pk.w[9] = WDesc{w[35], 512, 256, 512, 256, 307200};
  prep_weights_kernel<<<10, 256, 0, stream>>>(pk, wf16);

  fps_kernel<<<8, 1024, 0, stream>>>(pts, cent, cent_out);
  knn_kernel<<<8192, 256, 0, stream>>>(pts, cent, gxyz);

  MlpCfg c0;
  c0.k = 16; c0.colOff = 0; c0.gxyz = gxyz; c0.wf16 = wf16; c0.xout = x;
  c0.L[0] = MlpLayer{32,  32,  32,  0,     w[2],  w[3],  w[4]};
  c0.L[1] = MlpLayer{64,  64,  64,  1024,  w[6],  w[7],  w[8]};
  c0.L[2] = MlpLayer{96,  128, 125, 5120,  w[10], w[11], w[12]};
  mlp_kernel<<<8192, 128, 0, stream>>>(c0);

  MlpCfg c1;
  c1.k = 32; c1.colOff = 125; c1.gxyz = gxyz; c1.wf16 = wf16; c1.xout = x;
  c1.L[0] = MlpLayer{32,  64,  64,  17408, w[14], w[15], w[16]};
  c1.L[1] = MlpLayer{96,  96,  96,  19456, w[18], w[19], w[20]};
  c1.L[2] = MlpLayer{128, 128, 128, 28672, w[22], w[23], w[24]};
  mlp_kernel<<<8192, 128, 0, stream>>>(c1);

  finalize_features_kernel<<<1024, 256, 0, stream>>>(cent, x, xh, xhT);

  // q = x*Wq + bq ; k = x*Wk + bk  (stored f16 for the attention WMMAs)
  gemm_kernel<<<dim3(512, 2), 128, 0, stream>>>(xh, 256, wf16 + 45056, 256,
                                                w[26], 0, qh, (float*)nullptr);
  gemm_kernel<<<dim3(512, 2), 128, 0, stream>>>(xh, 256, wf16 + 110592, 256,
                                                w[28], 0, kh, (float*)nullptr);

  attn_kernel<<<1024, 128, 0, stream>>>(qh, kh, xhT, att);

  // mid = LN(att + x)
  ln_add_kernel<<<1024, 256, 0, stream>>>(att, x, w[29], w[30], mid, midh);

  // FFN 256 -> 512 (relu) -> 256
  gemm_kernel<<<dim3(512, 4), 128, 0, stream>>>(midh, 256, wf16 + 176128, 512,
                                                w[34], 1, h1h, (float*)nullptr);
  gemm_kernel<<<dim3(512, 2), 128, 0, stream>>>(h1h, 512, wf16 + 307200, 256,
                                                w[36], 0, (_Float16*)nullptr, att);

  // out = LN(h2 + mid)
  ln_add_kernel<<<1024, 256, 0, stream>>>(att, mid, w[31], w[32],
                                          feat_out, (_Float16*)nullptr);
}
